// BiLstmCRF_61881888800931
// MI455X (gfx1250) — matrix operand order repaired and checked by reading, not compiled, not run on hardware
//
#include <hip/hip_runtime.h>
#include <hip/hip_bf16.h>
#include <math.h>

typedef __attribute__((ext_vector_type(16))) _Float16 v16h;
typedef __attribute__((ext_vector_type(8)))  float    v8f;

namespace {
constexpr int Bc = 64, Tc = 256, Ec = 256, Hc = 256, Lc = 9;
constexpr int G4 = 4 * Hc;    // 1024 gate columns
constexpr int KK = Ec + Hc;   // 512 concat K-dim
constexpr long XH_ELEMS = (long)Bc * Tc * Ec;       // 4,194,304
constexpr long WT_ELEMS = (long)G4 * KK;            // 524,288 per direction
}

union F16x16 { v16h v; float4 q[2]; };

// Native-instruction nonlinearities for the hot loop (v_exp_f32 / v_rcp_f32).
__device__ __forceinline__ float fast_sig(float x) {
  return __builtin_amdgcn_rcpf(1.0f + __expf(-x));
}
__device__ __forceinline__ float fast_tanh(float x) {
  // 1 - 2/(e^{2x}+1); saturates to +/-1 correctly at both infinities.
  return 1.0f - 2.0f * __builtin_amdgcn_rcpf(__expf(2.0f * x) + 1.0f);
}

// ---------------------------------------------------------------------------
// 1) Embedding gather -> fp16 activations  x_h[b*T+t][e] = emb[inp[b,t]][e]
// ---------------------------------------------------------------------------
__global__ void k_gather(const int* __restrict__ inp, const float* __restrict__ emb,
                         _Float16* __restrict__ xh) {
  long i = (long)blockIdx.x * blockDim.x + threadIdx.x;
  if (i >= XH_ELEMS) return;
  int e  = (int)(i & (Ec - 1));
  int bt = (int)(i >> 8);
  int tok = inp[bt];
  xh[i] = (_Float16)emb[(size_t)tok * Ec + e];
}

// ---------------------------------------------------------------------------
// 2) Weight prep: WcatT[d][n][k] = fp16( k<E ? W_d[k][n] : U_d[k-E][n] )
//    (N-major, K-contiguous: makes B-fragment loads 16B-contiguous)
// ---------------------------------------------------------------------------
__global__ void k_prepw(const float* __restrict__ Wf, const float* __restrict__ Uf,
                        const float* __restrict__ Wb, const float* __restrict__ Ub,
                        _Float16* __restrict__ wT) {
  long i = (long)blockIdx.x * blockDim.x + threadIdx.x;
  if (i >= 2 * WT_ELEMS) return;
  int d = (int)(i / WT_ELEMS);
  long r = i - (long)d * WT_ELEMS;
  int n = (int)(r / KK);
  int k = (int)(r - (long)n * KK);
  const float* W = d ? Wb : Wf;
  const float* U = d ? Ub : Uf;
  float v = (k < Ec) ? W[(size_t)k * G4 + n] : U[(size_t)(k - Ec) * G4 + n];
  wT[i] = (_Float16)v;
}

// ---------------------------------------------------------------------------
// 3) Persistent LSTM: one 1024-thread workgroup per direction.
//    LDS holds the [64 x 512] fp16 activation matrix [x_t | h].
//    Wave w owns rows mt=w&3 (16 batch rows) and H-slices {2*(w>>2), +1};
//    for each slice it computes all 4 gate tiles so c/h stay in VGPRs.
//    An opaque per-step integer offset (not a laundered pointer!) blocks LICM
//    from hoisting weight loads while keeping the GLOBAL address space, so
//    B-fragments come in as global_load_b128 from L2 every step.
// ---------------------------------------------------------------------------
__global__ __launch_bounds__(1024) void k_lstm(
    const _Float16* __restrict__ xh, const _Float16* __restrict__ wT2,
    const float* __restrict__ bf, const float* __restrict__ bb,
    const int* __restrict__ inp,
    float* __restrict__ hf, float* __restrict__ hb) {
  extern __shared__ _Float16 Abuf[];   // [64][KK] = 64 KB
  const int d = blockIdx.x;
  const _Float16* wT = wT2 + (size_t)d * WT_ELEMS;
  const float* bias = d ? bb : bf;
  float* hs = d ? hb : hf;

  const int tid  = threadIdx.x;
  const int lane = tid & 31;           // wave32
  const int w    = tid >> 5;           // 0..31
  const int mt   = w & 3;              // batch row tile
  const int hg   = w >> 2;             // 0..7 -> H tiles {2hg, 2hg+1}
  const int half = lane >> 4;
  const int ln   = lane & 15;
  const int arow = mt * 16 + ln;       // A-matrix row this lane loads

  // zero the h half of the activation buffer (each thread: 16 halves)
  {
    int idx = tid * 16;
    int m = idx >> 8, e0 = idx & 255;
    float4* dst = (float4*)(void*)&Abuf[(size_t)m * KK + Ec + e0];
    float4 z = make_float4(0.f, 0.f, 0.f, 0.f);
    dst[0] = z; dst[1] = z;
  }

  // per-lane gate biases (column n = gt*256 + ht*16 + ln), loaded once
  float biasv[2][4];
#pragma unroll
  for (int g = 0; g < 2; ++g) {
    int ht = hg * 2 + g;
#pragma unroll
    for (int gt = 0; gt < 4; ++gt)
      biasv[g][gt] = bias[gt * Hc + ht * 16 + ln];
  }

  v8f cc[2], hreg[2];                  // cell/hidden state: persistent in VGPRs
#pragma unroll
  for (int g = 0; g < 2; ++g)
#pragma unroll
    for (int r = 0; r < 8; ++r) { cc[g][r] = 0.f; hreg[g][r] = 0.f; }

  for (int t = 0; t < Tc; ++t) {
    const int ta = d ? (Tc - 1 - t) : t;

    // Opaque zero offset, renewed every timestep: weight loads depend on it,
    // so they cannot be hoisted out of the t-loop (which previously caused a
    // mass spill to scratch), yet address-space inference still sees a
    // global-memory object -> global_load_b128, LOADcnt only.
    unsigned zoff = 0;
    asm volatile("" : "+s"(zoff));
    const _Float16* wTl = wT + zoff;

    __syncthreads();                   // h/x writes from previous step visible
    // stage x_t into the first E columns (each thread: 16 halves = 2x b128)
    {
      int idx = tid * 16;
      int m = idx >> 8, e0 = idx & 255;
      const float4* src = (const float4*)(const void*)(xh + ((size_t)m * Tc + ta) * Ec + e0);
      float4* dst = (float4*)(void*)&Abuf[(size_t)m * KK + e0];
      dst[0] = src[0]; dst[1] = src[1];
    }
    __syncthreads();

#pragma unroll
    for (int g = 0; g < 2; ++g) {
      const int ht = hg * 2 + g;
      v8f acc[4];
#pragma unroll
      for (int gt = 0; gt < 4; ++gt)
#pragma unroll
        for (int r = 0; r < 8; ++r) acc[gt][r] = biasv[g][gt];

#pragma unroll 2
      for (int kb = 0; kb < KK / 32; ++kb) {
        // A fragment: row `arow`, 16 contiguous K at kb*32 + 16*half
        F16x16 a;
        const _Float16* ap = &Abuf[(size_t)arow * KK + kb * 32 + half * 16];
        a.q[0] = *(const float4*)(const void*)(ap);
        a.q[1] = *(const float4*)(const void*)(ap + 8);
#pragma unroll
        for (int gt = 0; gt < 4; ++gt) {
          const int ntile = gt * 16 + ht;
          // B fragment: column ntile*16+ln, 16 contiguous K at kb*32 + 16*half
          F16x16 bm;
          const _Float16* bp = wTl + (size_t)(ntile * 16 + ln) * KK + kb * 32 + half * 16;
          bm.q[0] = *(const float4*)(const void*)(bp);
          bm.q[1] = *(const float4*)(const void*)(bp + 8);
          acc[gt] = __builtin_amdgcn_wmma_f32_16x16x32_f16(
              false, a.v, false, bm.v, (short)0, acc[gt], false, false);
        }
      }
      // gate nonlinearity + masked state update (all in registers, native
      // v_exp/v_rcp -- no divergent libm slow paths in the hot loop)
#pragma unroll
      for (int r = 0; r < 8; ++r) {
        int m = mt * 16 + half * 8 + r;     // C/D layout: M = r + 8*half
        bool mk = inp[m * Tc + ta] != 0;
        float iv = fast_sig(acc[0][r]);
        float fv = fast_sig(acc[1][r]);
        float gv = fast_tanh(acc[2][r]);
        float ov = fast_sig(acc[3][r]);
        float cn = fv * cc[g][r] + iv * gv;
        float hn = ov * fast_tanh(cn);
        if (mk) { cc[g][r] = cn; hreg[g][r] = hn; }
      }
    }
    __syncthreads();                   // all reads of old h complete
    // publish h_new: LDS (for next step's K-dim) + global (for decoder)
#pragma unroll
    for (int g = 0; g < 2; ++g) {
      const int ht = hg * 2 + g;
      const int col = ht * 16 + ln;
#pragma unroll
      for (int r = 0; r < 8; ++r) {
        int m = mt * 16 + half * 8 + r;
        float hv = hreg[g][r];
        Abuf[(size_t)m * KK + Ec + col] = (_Float16)hv;
        hs[((size_t)m * Tc + ta) * Hc + col] = hv;
      }
    }
  }
}

// ---------------------------------------------------------------------------
// 4) Decoder: pot[b,t,l] = [hf|hb] @ Wd + bd   (L=9, trivial VALU)
// ---------------------------------------------------------------------------
__global__ void k_dec(const float* __restrict__ hf, const float* __restrict__ hb,
                      const float* __restrict__ Wd, const float* __restrict__ bd,
                      float* __restrict__ pot) {
  int i = blockIdx.x * blockDim.x + threadIdx.x;
  if (i >= Bc * Tc * Lc) return;
  int bt = i / Lc, l = i - bt * Lc;
  const float* hfp = hf + (size_t)bt * Hc;
  const float* hbp = hb + (size_t)bt * Hc;
  float s = bd[l];
  for (int k = 0; k < Hc; ++k) s += hfp[k] * Wd[(size_t)k * Lc + l];
  for (int k = 0; k < Hc; ++k) s += hbp[k] * Wd[(size_t)(Hc + k) * Lc + l];
  pot[i] = s;
}

// ---------------------------------------------------------------------------
// 5) CRF log-likelihood + loss: one thread per batch element (L=9)
//    (precise libm here: cold code, logsumexp accuracy matters)
// ---------------------------------------------------------------------------
__global__ void k_crf(const float* __restrict__ pot, const int* __restrict__ inp,
                      const int* __restrict__ tags, const float* __restrict__ trans,
                      float* __restrict__ oseq, float* __restrict__ oloss) {
  __shared__ float tr[Lc * Lc];
  __shared__ float red[Bc];
  int b = threadIdx.x;
  for (int i = b; i < Lc * Lc; i += Bc) tr[i] = trans[i];
  __syncthreads();
  const float* pb = pot + (size_t)b * Tc * Lc;
  const int* tg = tags + (size_t)b * Tc;
  const int* ib = inp + (size_t)b * Tc;
  int len = 0;
  for (int t = 0; t < Tc; ++t) len += (ib[t] != 0);
  float unary = 0.f;
  for (int t = 0; t < Tc; ++t) if (t < len) unary += pb[t * Lc + tg[t]];
  float binary = 0.f;
  for (int t = 0; t + 1 < Tc; ++t) if (t < len - 1) binary += tr[tg[t] * Lc + tg[t + 1]];
  float alpha[Lc];
  for (int j = 0; j < Lc; ++j) alpha[j] = pb[j];
  for (int t = 1; t < Tc; ++t) {
    if (t < len) {
      float na[Lc];
      for (int j = 0; j < Lc; ++j) {
        float mx = -1e30f;
        for (int i = 0; i < Lc; ++i) mx = fmaxf(mx, alpha[i] + tr[i * Lc + j]);
        float s = 0.f;
        for (int i = 0; i < Lc; ++i) s += expf(alpha[i] + tr[i * Lc + j] - mx);
        na[j] = pb[t * Lc + j] + mx + logf(s);
      }
      for (int j = 0; j < Lc; ++j) alpha[j] = na[j];
    }
  }
  float mx = -1e30f;
  for (int j = 0; j < Lc; ++j) mx = fmaxf(mx, alpha[j]);
  float s = 0.f;
  for (int j = 0; j < Lc; ++j) s += expf(alpha[j] - mx);
  float log_norm = mx + logf(s);
  float ll = unary + binary - log_norm;
  oseq[b] = (float)len;
  red[b] = -(ll / (float)len) * (1.0f / Bc);
  __syncthreads();
  if (b == 0) {
    float acc = 0.f;
    for (int i = 0; i < Bc; ++i) acc += red[i];
    oloss[0] = acc;
  }
}

// ---------------------------------------------------------------------------
extern "C" void kernel_launch(void* const* d_in, const int* in_sizes, int n_in,
                              void* d_out, int out_size, void* d_ws, size_t ws_size,
                              hipStream_t stream) {
  (void)in_sizes; (void)n_in; (void)out_size; (void)ws_size;
  const int*   inp   = (const int*)  d_in[0];
  const int*   tags  = (const int*)  d_in[1];
  const float* emb   = (const float*)d_in[2];
  const float* Wf    = (const float*)d_in[3];
  const float* Uf    = (const float*)d_in[4];
  const float* bf    = (const float*)d_in[5];
  const float* Wb    = (const float*)d_in[6];
  const float* Ub    = (const float*)d_in[7];
  const float* bb    = (const float*)d_in[8];
  const float* Wd    = (const float*)d_in[9];
  const float* bd    = (const float*)d_in[10];
  const float* trans = (const float*)d_in[11];

  char* ws = (char*)d_ws;
  _Float16* xh = (_Float16*)(ws);                                // 8,388,608 B
  _Float16* wT = (_Float16*)(ws + 8388608);                      // 2,097,152 B
  float*    hf = (float*)   (ws + 8388608 + 2097152);            // 16,777,216 B
  float*    hb = (float*)   (ws + 8388608 + 2097152 + 16777216); // 16,777,216 B

  const int thr = 256;
  k_gather<<<(unsigned)((XH_ELEMS + thr - 1) / thr), thr, 0, stream>>>(inp, emb, xh);
  k_prepw<<<(unsigned)((2 * WT_ELEMS + thr - 1) / thr), thr, 0, stream>>>(Wf, Uf, Wb, Ub, wT);
  k_lstm<<<2, 1024, (size_t)Bc * KK * sizeof(_Float16), stream>>>(xh, wT, bf, bb, inp, hf, hb);
  k_dec<<<(unsigned)((Bc * Tc * Lc + thr - 1) / thr), thr, 0, stream>>>(hf, hb, Wd, bd, (float*)d_out);
  k_crf<<<1, Bc, 0, stream>>>((const float*)d_out, inp, tags, trans,
                              (float*)d_out + (size_t)Bc * Tc * Lc,
                              (float*)d_out + (size_t)Bc * Tc * Lc + Bc);
}
